// LSTMBinaryClassifier_68762426409597
// MI455X (gfx1250) — matrix operand order repaired and checked by reading, not compiled, not run on hardware
//
#include <hip/hip_runtime.h>

typedef __attribute__((ext_vector_type(16))) _Float16 v16h;
typedef __attribute__((ext_vector_type(8)))  _Float16 v8h;
typedef __attribute__((ext_vector_type(8)))  float    v8f;

#define HID    64
#define NGATE  4
#define CHUNK  128
#define BTILE  16

union V16 { v16h v; v8h h[2]; };

__device__ __forceinline__ float fast_sigmoid(float x) {
    // sigma(x) = 1 / (1 + e^-x)   -> v_exp_f32 + v_rcp_f32
    return __builtin_amdgcn_rcpf(1.0f + __expf(-x));
}
__device__ __forceinline__ float fast_tanh(float x) {
    // tanh(x) = 2*sigma(2x) - 1
    return 2.0f * __builtin_amdgcn_rcpf(1.0f + __expf(-2.0f * x)) - 1.0f;
}

__global__ __launch_bounds__(128) void lstm_wmma_kernel(
    const float* __restrict__ x,     const float* __restrict__ W_ih,
    const float* __restrict__ W_hh,  const float* __restrict__ b_ih,
    const float* __restrict__ b_hh,  const float* __restrict__ W_lin,
    const float* __restrict__ b_lin, float* __restrict__ out,
    int B, int T)
{
    __shared__ __align__(16) float    xbuf[2][BTILE * CHUNK];   // 16 KB
    __shared__ __align__(16) _Float16 hbuf[2][BTILE * HID];     //  4 KB

    const int tid  = threadIdx.x;
    const int wave = tid >> 5;
    const int lane = tid & 31;
    const int n    = lane & 15;   // N column within 16-wide tile (also A-row M)
    const int hi   = lane >> 4;   // half-wave select
    const int b0   = blockIdx.x * BTILE;

    // ---- one-time: W_hh -> WMMA B fragments (per-wave, kept in VGPRs) ----
    // gates = h @ W_hh^T  =>  B[k][n] = W_hh[nglob][k]
    // B fragment k order used here (A matches): lane n, lanes0-15 hold K=kbase..kbase+15,
    // lanes16-31 hold K=kbase+16..kbase+31 (two linear runs of 16).
    v16h WB[NGATE][2];
    float wih[NGATE], bsum[NGATE];
    #pragma unroll
    for (int G = 0; G < NGATE; ++G) {
        const int nglob = G * HID + wave * 16 + n;
        wih[G]  = W_ih[nglob];
        bsum[G] = b_ih[nglob] + b_hh[nglob];
        #pragma unroll
        for (int k = 0; k < 2; ++k) {
            const int kbase = 32 * k + hi * 16;
            v16h wb;
            #pragma unroll
            for (int e = 0; e < 16; ++e)
                wb[e] = (_Float16)W_hh[nglob * HID + kbase + e];
            WB[G][k] = wb;
        }
    }

    // ---- recurrent state ----
    v16h A[2];                 // h as A fragments; h0 = 0
    #pragma unroll
    for (int k = 0; k < 2; ++k)
        #pragma unroll
        for (int e = 0; e < 16; ++e) A[k][e] = (_Float16)0.0f;
    v8f c = {};                // cell state, fp32, accumulator layout

    const int nchunks = (T + CHUNK - 1) / CHUNK;
    for (int ch = 0; ch < nchunks; ++ch) {
        const int t0 = ch * CHUNK;
        const int cb = ch & 1;

        // stage x[b0..b0+15][t0..t0+CHUNK) into LDS, coalesced float4 loads
        #pragma unroll
        for (int i = 0; i < (BTILE * CHUNK / 4) / 128; ++i) {
            const int idx  = tid + i * 128;
            const int row  = idx >> 5;          // 32 float4 per row
            const int col4 = idx & 31;
            const int tg   = t0 + col4 * 4;
            float4 v = make_float4(0.f, 0.f, 0.f, 0.f);
            if (tg + 3 < T)
                v = *(const float4*)(x + (size_t)(b0 + row) * T + tg);
            *(float4*)&xbuf[cb][row * CHUNK + col4 * 4] = v;
            if (t0 + CHUNK < T)   // pull next chunk toward L2/L0 early
                __builtin_prefetch(x + (size_t)(b0 + row) * T + tg + CHUNK, 0, 0);
        }
        __syncthreads();

        const int steps = (T - t0 < CHUNK) ? (T - t0) : CHUNK;
        for (int tc = 0; tc < steps; ++tc) {
            const int nb = (t0 + tc + 1) & 1;    // h buffer to write

            // x scalar for batch row M = r + 8*hi (LDS broadcast)
            float xs[8];
            #pragma unroll
            for (int r = 0; r < 8; ++r)
                xs[r] = xbuf[cb][(r + 8 * hi) * CHUNK + tc];

            // gates = (x*W_ih + b_ih + b_hh) + h @ W_hh^T
            v8f acc[NGATE];
            #pragma unroll
            for (int G = 0; G < NGATE; ++G) {
                v8f a0;
                #pragma unroll
                for (int r = 0; r < 8; ++r)
                    a0[r] = fmaf(xs[r], wih[G], bsum[G]);
                #pragma unroll
                for (int k = 0; k < 2; ++k)
                    a0 = __builtin_amdgcn_wmma_f32_16x16x32_f16(
                        false, A[k], false, WB[G][k], (short)0, a0,
                        false, false);
                acc[G] = a0;
            }

            // LSTM cell update (wave-local, fp32) + write f16 h tile to LDS
            #pragma unroll
            for (int r = 0; r < 8; ++r) {
                const float ig = fast_sigmoid(acc[0][r]);
                const float fg = fast_sigmoid(acc[1][r]);
                const float gg = fast_tanh  (acc[2][r]);
                const float og = fast_sigmoid(acc[3][r]);
                const float cn = fg * c[r] + ig * gg;
                c[r] = cn;
                const float hn = og * fast_tanh(cn);
                const int   m  = r + 8 * hi;
                hbuf[nb][m * HID + wave * 16 + n] = (_Float16)hn;
            }
            __syncthreads();   // single barrier per timestep (double-buffered)

            // reload h as A fragments: two ds_load_b128 per K-tile
            #pragma unroll
            for (int k = 0; k < 2; ++k) {
                const _Float16* p = &hbuf[nb][n * HID + 32 * k + hi * 16];
                V16 t16;
                t16.h[0] = *(const v8h*)(p);        // K = kbase .. kbase+7
                t16.h[1] = *(const v8h*)(p + 8);    // K = kbase+8 .. +15
                A[k] = t16.v;
            }
        }
    }

    // ---- final: logits = h_last @ W_lin^T + b_lin ; log_softmax ----
    if (tid < BTILE) {
        const _Float16* hrow = &hbuf[T & 1][tid * HID];
        float l[3];
        #pragma unroll
        for (int cc = 0; cc < 3; ++cc) {
            float s = b_lin[cc];
            #pragma unroll 8
            for (int k2 = 0; k2 < HID; ++k2)
                s = fmaf((float)hrow[k2], W_lin[cc * HID + k2], s);
            l[cc] = s;
        }
        const float mx  = fmaxf(l[0], fmaxf(l[1], l[2]));
        const float se  = __expf(l[0] - mx) + __expf(l[1] - mx) + __expf(l[2] - mx);
        const float lse = mx + __logf(se);
        #pragma unroll
        for (int cc = 0; cc < 3; ++cc)
            out[(size_t)(b0 + tid) * 3 + cc] = l[cc] - lse;
    }
}

extern "C" void kernel_launch(void* const* d_in, const int* in_sizes, int n_in,
                              void* d_out, int out_size, void* d_ws, size_t ws_size,
                              hipStream_t stream) {
    const float* x     = (const float*)d_in[0];
    const float* W_ih  = (const float*)d_in[1];
    const float* W_hh  = (const float*)d_in[2];
    const float* b_ih  = (const float*)d_in[3];
    const float* b_hh  = (const float*)d_in[4];
    const float* W_lin = (const float*)d_in[5];
    const float* b_lin = (const float*)d_in[6];
    float* out = (float*)d_out;

    const int B = out_size / 3;           // 2048
    const int T = in_sizes[0] / B;        // 2048 (x is [B,T,1])

    dim3 grid(B / BTILE);                 // 128 persistent workgroups
    dim3 block(128);                      // 4 waves (wave32)
    lstm_wmma_kernel<<<grid, block, 0, stream>>>(
        x, W_ih, W_hh, b_ih, b_hh, W_lin, b_lin, out, B, T);
}
